// BasicNeRF2d_50861002719868
// MI455X (gfx1250) — matrix operand order repaired and checked by reading, not compiled, not run on hardware
//
#include <hip/hip_runtime.h>
#include <hip/hip_bf16.h>

typedef __attribute__((ext_vector_type(16))) _Float16 v16h;
typedef __attribute__((ext_vector_type(8)))  _Float16 v8h;
typedef __attribute__((ext_vector_type(8)))  float    v8f;

#define C_DIM      256
#define BLK_THR    128           // 4 waves
#define WAVES      4
#define ROWS_W     32            // rows (points) per wave  -> 2x B-fragment reuse
#define TILE_PTS   (WAVES * ROWS_W)       // 128 points per block-iteration
#define KTILES     8             // 256 / 32
#define NTILES     16            // 256 / 16
#define WT_HALFS   (C_DIM * C_DIM)        // 65536 halves = 128KB per matrix
#define SCR_HALFS  (ROWS_W * C_DIM)       // 8192 halves = 16KB per wave
#define SCR_TOTAL  (WAVES * SCR_HALFS)    // 32768 halves = 64KB

union FragU { v16h v; v8h h[2]; };

struct F3 { float x, y, z; };            // packed 12B -> global_load_b96

__global__ __launch_bounds__(BLK_THR)
void nerf2d_wmma_kernel(const float* __restrict__ uv,
                        const float* __restrict__ W_in,  const float* __restrict__ b_in,
                        const float* __restrict__ W_h0,  const float* __restrict__ b_h0,
                        const float* __restrict__ W_h1,  const float* __restrict__ b_h1,
                        const float* __restrict__ W_out, const float* __restrict__ b_out,
                        float* __restrict__ out,
                        int pointsPerBlock)
{
    extern __shared__ _Float16 lds[];
    // Scratch FIRST: keeps all activation addresses < 64KB so row offsets
    // fold into the 16-bit DS offset immediate (no per-store v_add chains).
    _Float16* scratch = lds;                         // 4 waves x [32][256] f16
    _Float16* wt0     = lds + SCR_TOTAL;             // [n][k] transposed W_h0
    _Float16* wt1     = lds + SCR_TOTAL + WT_HALFS;  // [n][k] transposed W_h1

    const int tid      = threadIdx.x;
    const int wave     = tid >> 5;
    const int lane     = tid & 31;
    const int lane15   = lane & 15;
    const int laneHalf = lane >> 4;                  // 0: low half, 1: high half

    // ---- Stage hidden-layer weights into LDS as f16, transposed [n][k] ----
    for (int i = tid; i < WT_HALFS; i += BLK_THR) {
        const int k = i >> 8;
        const int n = i & 255;
        wt0[n * C_DIM + k] = (_Float16)W_h0[i];      // W_h0 row-major [k][n]
        wt1[n * C_DIM + k] = (_Float16)W_h1[i];
    }
    __syncthreads();

    // ---- Preload W_in columns for this lane (layer-1 is VALU, K=4) ----
    float w0[8], w1[8], w2[8], w3[8], bb[8];
#pragma unroll
    for (int j = 0; j < 8; ++j) {
        const int c = lane * 8 + j;
        w0[j] = W_in[0 * C_DIM + c];
        w1[j] = W_in[1 * C_DIM + c];
        w2[j] = W_in[2 * C_DIM + c];
        w3[j] = W_in[3 * C_DIM + c];
        bb[j] = b_in[c];
    }

    _Float16* sc = scratch + wave * SCR_HALFS;       // this wave's [32][256] tile
    const long base0 = (long)blockIdx.x * (long)pointsPerBlock;
    const int  iters = pointsPerBlock / TILE_PTS;
    const float2* uv2 = (const float2*)uv;
    const F3* wout3 = (const F3*)W_out;

    for (int it = 0; it < iters; ++it) {
        const long rowBase = base0 + (long)it * TILE_PTS + wave * ROWS_W;

        // ========== Layer 1: enc[32,4] @ W_in[4,256] -> scratch f16 ==========
        const float2 p2 = uv2[rowBase + lane];       // one point per lane
        const float cu = __cosf(p2.x), cv = __cosf(p2.y);
        const float su = __sinf(p2.x), sv = __sinf(p2.y);

#pragma unroll
        for (int m = 0; m < ROWS_W; ++m) {
            const float e0 = __shfl(cu, m, 32);
            const float e1 = __shfl(cv, m, 32);
            const float e2 = __shfl(su, m, 32);
            const float e3 = __shfl(sv, m, 32);
            v8h row;
#pragma unroll
            for (int j = 0; j < 8; ++j) {
                float a = bb[j];
                a = __builtin_fmaf(e0, w0[j], a);
                a = __builtin_fmaf(e1, w1[j], a);
                a = __builtin_fmaf(e2, w2[j], a);
                a = __builtin_fmaf(e3, w3[j], a);
                row[j] = (_Float16)(a > 0.0f ? a : 0.0f);
            }
            *(v8h*)(sc + m * C_DIM + lane * 8) = row;   // ds_store_b128
        }

        // ========== Hidden layers: two [32,256] @ [256,256] WMMA GEMMs =======
#pragma unroll
        for (int layer = 0; layer < 2; ++layer) {
            const _Float16* wt = layer ? wt1 : wt0;
            const float*    bs = layer ? b_h1 : b_h0;

            // Preload this layer's per-lane biases (hoists global loads +
            // waitcnt out of the N-tile loop; live range = this layer only).
            float bsr[NTILES];
#pragma unroll
            for (int nb = 0; nb < NTILES; ++nb) bsr[nb] = bs[nb * 16 + lane15];

            asm volatile("s_wait_dscnt 0" ::: "memory");

            // Two A-fragment sets (rows 0-15 and 16-31), 16x32 f16 each.
            // Lane<16 holds K 0-7 & 16-23 of its row, lane>=16 holds
            // K 8-15 & 24-31 (CDNA5 WMMA A layout).
            v16h A0[KTILES], A1[KTILES];
#pragma unroll
            for (int kb = 0; kb < KTILES; ++kb) {
                const _Float16* p0 = sc + lane15 * C_DIM + kb * 32 + laneHalf * 8;
                const _Float16* p1 = p0 + 16 * C_DIM;
                FragU f0, f1;
                f0.h[0] = *(const v8h*)(p0);
                f0.h[1] = *(const v8h*)(p0 + 16);
                f1.h[0] = *(const v8h*)(p1);
                f1.h[1] = *(const v8h*)(p1 + 16);
                A0[kb] = f0.v;
                A1[kb] = f1.v;
            }

            for (int nb = 0; nb < NTILES; ++nb) {
                const int ncol = nb * 16 + lane15;
                const float bias = bsr[nb];
                v8f acc0, acc1;
#pragma unroll
                for (int r = 0; r < 8; ++r) { acc0[r] = bias; acc1[r] = bias; }

#pragma unroll
                for (int kb = 0; kb < KTILES; ++kb) {
                    // B fragment loaded ONCE, feeds TWO WMMAs (row halves)
                    const _Float16* pb = wt + ncol * C_DIM + kb * 32 + laneHalf * 8;
                    FragU fb;
                    fb.h[0] = *(const v8h*)(pb);
                    fb.h[1] = *(const v8h*)(pb + 16);
                    const v16h B = fb.v;
                    acc0 = __builtin_amdgcn_wmma_f32_16x16x32_f16(
                            false, A0[kb], false, B, (short)0, acc0, false, false);
                    acc1 = __builtin_amdgcn_wmma_f32_16x16x32_f16(
                            false, A1[kb], false, B, (short)0, acc1, false, false);
                }

                // ReLU + f16, scatter to scratch (D layout: col=lane15,
                // rows laneHalf*8 + r; second set offset by 16 rows)
                _Float16* st = sc + ncol;            // one add per nb; row
#pragma unroll                                       // offsets are immediates
                for (int r = 0; r < 8; ++r) {
                    const int m = laneHalf * 8 + r;
                    const float r0 = acc0[r];
                    const float r1 = acc1[r];
                    st[m * C_DIM]        = (_Float16)(r0 > 0.0f ? r0 : 0.0f);
                    st[(m + 16) * C_DIM] = (_Float16)(r1 > 0.0f ? r1 : 0.0f);
                }
            }
        }

        // ========== Output layer: [32,256] @ [256,3] + sigmoid ===============
        asm volatile("s_wait_dscnt 0" ::: "memory");
        // Each lane owns one point: full 256-channel dot, no reduction needed.
        float o0 = b_out[0], o1 = b_out[1], o2 = b_out[2];
        const _Float16* hp = sc + lane * C_DIM;
        for (int c8 = 0; c8 < 32; ++c8) {
            const v8h hv = *(const v8h*)(hp + c8 * 8);
#pragma unroll
            for (int j = 0; j < 8; ++j) {
                const float h = (float)hv[j];
                const F3 w = wout3[c8 * 8 + j];      // global_load_b96
                o0 = __builtin_fmaf(h, w.x, o0);
                o1 = __builtin_fmaf(h, w.y, o1);
                o2 = __builtin_fmaf(h, w.z, o2);
            }
        }

        const long p = rowBase + lane;
        out[p * 3 + 0] = 1.0f / (1.0f + __expf(-o0));
        out[p * 3 + 1] = 1.0f / (1.0f + __expf(-o1));
        out[p * 3 + 2] = 1.0f / (1.0f + __expf(-o2));
    }
}

extern "C" void kernel_launch(void* const* d_in, const int* in_sizes, int n_in,
                              void* d_out, int out_size, void* d_ws, size_t ws_size,
                              hipStream_t stream) {
    const float* uv    = (const float*)d_in[0];
    const float* W_in  = (const float*)d_in[1];
    const float* b_in  = (const float*)d_in[2];
    const float* W_h0  = (const float*)d_in[3];
    const float* b_h0  = (const float*)d_in[4];
    const float* W_h1  = (const float*)d_in[5];
    const float* b_h1  = (const float*)d_in[6];
    const float* W_out = (const float*)d_in[7];
    const float* b_out = (const float*)d_in[8];
    float* out = (float*)d_out;

    const int nPoints = in_sizes[0] / 2;             // 524288
    const int nBlocks = 512;                         // 1024 points/block, 8 iters
    const int ppb     = nPoints / nBlocks;
    // 64KB activation scratch + 2 x 128KB f16 weights = 320KB LDS (WGP max)
    const size_t ldsBytes = (size_t)(SCR_TOTAL + 2 * WT_HALFS) * sizeof(_Float16);

    nerf2d_wmma_kernel<<<nBlocks, BLK_THR, ldsBytes, stream>>>(
        uv, W_in, b_in, W_h0, b_h0, W_h1, b_h1, W_out, b_out, out, ppb);
}
